// Itloss_18433999635078
// MI455X (gfx1250) — matrix operand order repaired and checked by reading, not compiled, not run on hardware
//
#include <hip/hip_runtime.h>
#include <hip/hip_bf16.h>
#include <math.h>

typedef __attribute__((ext_vector_type(16))) _Float16 v16h;
typedef __attribute__((ext_vector_type(8)))  _Float16 v8h;
typedef __attribute__((ext_vector_type(8)))  float    v8f;

union V16 { v16h v; v8h h[2]; };

#define NB 32768
#define NC 1000
#define NCP 1024      // padded to 64 tiles of 16 so cross-tile prefetch is safe
#define NT 63         // live column tiles (63*16 = 1008 >= 1000)
#define ND 512
#define INV_T 10.0f   // 1/TEMPERATURE

__device__ __forceinline__ V16 load_b16(const _Float16* p, int kk) {
  V16 b;
  b.h[0] = *(const v8h*)(p + kk * 32);
  b.h[1] = *(const v8h*)(p + kk * 32 + 16);
  return b;
}

// ---------------------------------------------------------------------------
// Kernel 1: normalize prototype rows (f32 -> f16), zero-pad rows 1000..1023,
// zero the 3 scalar accumulators (ws[0]=ce, ws[1]=sim, ws[2]=gram-expsum).
// ---------------------------------------------------------------------------
__global__ __launch_bounds__(128) void norm_proto_kernel(
    const float* __restrict__ proto_in, _Float16* __restrict__ proto16,
    float* __restrict__ accs)
{
  const int row = blockIdx.x;
  const int tid = threadIdx.x;
  if (row == 0 && tid < 3) accs[tid] = 0.0f;

  _Float16* dst = proto16 + (size_t)row * ND;
  if (row >= NC) {
    #pragma unroll
    for (int i = 0; i < 4; ++i) dst[tid + i * 128] = (_Float16)0.0f;
    return;
  }
  __shared__ float red[128];
  const float* src = proto_in + (size_t)row * ND;
  float v[4];
  float ss = 0.0f;
  #pragma unroll
  for (int i = 0; i < 4; ++i) { v[i] = src[tid + i * 128]; ss += v[i] * v[i]; }
  red[tid] = ss;
  __syncthreads();
  for (int d = 64; d > 0; d >>= 1) {
    if (tid < d) red[tid] += red[tid + d];
    __syncthreads();
  }
  const float inv = 1.0f / fmaxf(sqrtf(red[0]), 1e-12f);
  #pragma unroll
  for (int i = 0; i < 4; ++i) dst[tid + i * 128] = (_Float16)(v[i] * inv);
}

// ---------------------------------------------------------------------------
// Kernel 2: cross entropy, one wave32 per row, online logsumexp. Streaming,
// HBM-bandwidth bound (131 MB of logits).
// ---------------------------------------------------------------------------
__global__ __launch_bounds__(256) void ce_kernel(
    const float* __restrict__ logits, const int* __restrict__ labels,
    float* __restrict__ acc_ce)
{
  const int wave = threadIdx.x >> 5;
  const int lane = threadIdx.x & 31;
  const int row  = blockIdx.x * 8 + wave;
  const float* lg = logits + (size_t)row * NC;
  const int lab = labels[row];

  float m = -3.4e38f, s = 0.0f, t = 0.0f;
  for (int c = lane; c < NC; c += 32) {
    float x = lg[c];
    if (c == lab) t = x;
    if (x > m) { s = s * __expf(m - x) + 1.0f; m = x; }
    else       { s += __expf(x - m); }
  }
  #pragma unroll
  for (int d = 1; d < 32; d <<= 1) {
    float mo = __shfl_xor(m, d, 32);
    float so = __shfl_xor(s, d, 32);
    t += __shfl_xor(t, d, 32);
    float mn = fmaxf(m, mo);
    s = s * __expf(m - mn) + so * __expf(mo - mn);
    m = mn;
  }
  if (lane == 0) atomicAdd(acc_ce, -(t - m - __logf(s)));
}

// ---------------------------------------------------------------------------
// Kernel 3: fused f16 WMMA GEMM (features @ protos^T) + online softmax.
// 2 waves / block; each wave owns 16 rows. A (16x512 f16) lives in VGPRs for
// the whole column sweep; the B stream is software-pipelined depth-2 across
// tile boundaries so every WMMA runs with 4 b128 loads in flight
// (s_wait_loadcnt 4 instead of 0 -> no per-WMMA L2 latency exposure).
// ---------------------------------------------------------------------------
__global__ __launch_bounds__(64) void gemm_softmax_kernel(
    const float* __restrict__ feat, const _Float16* __restrict__ proto16,
    const int* __restrict__ labels, float* __restrict__ acc_sim)
{
  __shared__ _Float16 smem[2][16 * ND];   // 32 KB
  const int wave = threadIdx.x >> 5;
  const int lane = threadIdx.x & 31;
  const int rowbase = (blockIdx.x * 2 + wave) * 16;

  const int halfsel = lane >> 4;    // 0: K-low half, 1: K-high half
  const int nidx    = lane & 15;    // N column within tile / row stripe index

  // B base pointer for this lane (tile ct adds ct*16*ND)
  const _Float16* browBase = proto16 + (size_t)nidx * ND + halfsel * 8;

  // prime the depth-2 B pipeline (tile 0, kk = 0,1) while A is being staged
  V16 bA = load_b16(browBase, 0);
  V16 bB = load_b16(browBase, 1);

  // stage A: 16 rows x 512 f32 -> f16 in LDS (2048 float4 per wave)
  {
    const float4* src = (const float4*)(feat + (size_t)rowbase * ND);
    for (int i = lane; i < 2048; i += 32) {
      float4 f = src[i];
      _Float16* d = &smem[wave][0] + i * 4;
      d[0] = (_Float16)f.x; d[1] = (_Float16)f.y;
      d[2] = (_Float16)f.z; d[3] = (_Float16)f.w;
    }
  }
  __syncthreads();

  // hoist all 16 A fragments (16x32 f16 each) into VGPRs: 128 VGPRs
  V16 afrag[16];
  {
    const _Float16* arow = &smem[wave][0] + (size_t)nidx * ND + halfsel * 8;
    #pragma unroll
    for (int kk = 0; kk < 16; ++kk) {
      afrag[kk].h[0] = *(const v8h*)(arow + kk * 32);        // K {0..7}/{8..15}
      afrag[kk].h[1] = *(const v8h*)(arow + kk * 32 + 16);   // K {16..23}/{24..31}
    }
  }

  float m[8], s[8], tgt[8];
  int lab[8];
  #pragma unroll
  for (int r = 0; r < 8; ++r) {
    m[r] = -3.4e38f; s[r] = 0.0f; tgt[r] = 0.0f;
    lab[r] = labels[rowbase + r + 8 * halfsel];
  }

  for (int ct = 0; ct < NT; ++ct) {
    const _Float16* brow   = browBase + (size_t)ct * 16 * ND;
    const _Float16* brow_n = brow + 16 * ND;   // tile ct+1 (padded rows for ct=62)
    v8f acc = {};
    #pragma unroll
    for (int kk = 0; kk < 16; ++kk) {
      // prefetch 2 steps ahead, rolling over into the next tile's kk=0,1
      V16 bC = (kk + 2 < 16) ? load_b16(brow, kk + 2)
                             : load_b16(brow_n, kk - 14);
      acc = __builtin_amdgcn_wmma_f32_16x16x32_f16(
          false, afrag[kk].v, false, bA.v, (short)0, acc, false, false);
      bA = bB; bB = bC;
    }
    const int col = ct * 16 + nidx;
    if (col < NC) {
      #pragma unroll
      for (int r = 0; r < 8; ++r) {
        float x = acc[r] * INV_T;
        if (lab[r] == col) tgt[r] = x;
        if (x > m[r]) { s[r] = s[r] * __expf(m[r] - x) + 1.0f; m[r] = x; }
        else          { s[r] += __expf(x - m[r]); }
      }
    }
  }

  // merge across the 16 lanes holding different N columns of each row
  #pragma unroll
  for (int d = 1; d < 16; d <<= 1) {
    #pragma unroll
    for (int r = 0; r < 8; ++r) {
      float mo = __shfl_xor(m[r], d, 32);
      float so = __shfl_xor(s[r], d, 32);
      tgt[r] += __shfl_xor(tgt[r], d, 32);
      float mn = fmaxf(m[r], mo);
      s[r] = s[r] * __expf(m[r] - mn) + so * __expf(mo - mn);
      m[r] = mn;
    }
  }
  if (nidx == 0) {   // lanes 0 (rows 0..7) and 16 (rows 8..15)
    float part = 0.0f;
    #pragma unroll
    for (int r = 0; r < 8; ++r) part += -(tgt[r] - m[r] - __logf(s[r]));
    atomicAdd(acc_sim, part);
  }
}

// ---------------------------------------------------------------------------
// Kernel 4: gram = protos16 @ protos16^T, sum exp over off-diagonal entries.
// One wave per 16x16 tile, 63x63 tiles, 16 WMMAs each (~1 GFLOP total).
// ---------------------------------------------------------------------------
__global__ __launch_bounds__(32) void gram_kernel(
    const _Float16* __restrict__ proto16, float* __restrict__ acc_txt)
{
  const int rt = blockIdx.x / NT;
  const int ctile = blockIdx.x % NT;
  const int lane = threadIdx.x;
  const int halfsel = lane >> 4;
  const int nidx = lane & 15;

  const _Float16* arow = proto16 + (size_t)(rt * 16 + nidx) * ND + halfsel * 8;
  const _Float16* brow = proto16 + (size_t)(ctile * 16 + nidx) * ND + halfsel * 8;

  v8f acc = {};
  #pragma unroll
  for (int kk = 0; kk < 16; ++kk) {
    const int kb = kk * 32;
    V16 a, b;
    a.h[0] = *(const v8h*)(arow + kb);
    a.h[1] = *(const v8h*)(arow + kb + 16);
    b.h[0] = *(const v8h*)(brow + kb);
    b.h[1] = *(const v8h*)(brow + kb + 16);
    acc = __builtin_amdgcn_wmma_f32_16x16x32_f16(
        false, a.v, false, b.v, (short)0, acc, false, false);
  }

  float local = 0.0f;
  const int j = ctile * 16 + nidx;
  #pragma unroll
  for (int r = 0; r < 8; ++r) {
    const int i = rt * 16 + r + 8 * halfsel;
    if (i < NC && j < NC && i != j) local += __expf(acc[r]);
  }
  #pragma unroll
  for (int d = 1; d < 32; d <<= 1) local += __shfl_xor(local, d, 32);
  if (lane == 0) atomicAdd(acc_txt, local);
}

// ---------------------------------------------------------------------------
// Kernel 5: combine. out = ALPHA*sim/B + BETA*ce/B + log(expsum)/C
// ---------------------------------------------------------------------------
__global__ void finalize_kernel(const float* __restrict__ accs,
                                float* __restrict__ out)
{
  out[0] = accs[1] * (1.0f / (float)NB)      // ALPHA = 1
         + accs[0] * (1.0f / (float)NB)      // BETA  = 1
         + logf(accs[2]) * (1.0f / (float)NC);
}

// ---------------------------------------------------------------------------
extern "C" void kernel_launch(void* const* d_in, const int* in_sizes, int n_in,
                              void* d_out, int out_size, void* d_ws, size_t ws_size,
                              hipStream_t stream)
{
  const float* feat    = (const float*)d_in[0];   // [B, D] f32
  const float* logits  = (const float*)d_in[1];   // [B, C] f32
  const int*   labels  = (const int*)d_in[2];     // [B] int
  const float* proto   = (const float*)d_in[3];   // [C, D] f32

  float*     accs    = (float*)d_ws;                         // [0]=ce [1]=sim [2]=gram
  _Float16*  proto16 = (_Float16*)((char*)d_ws + 1024);      // [1024, 512] f16
  float*     out     = (float*)d_out;

  norm_proto_kernel<<<NCP, 128, 0, stream>>>(proto, proto16, accs);
  ce_kernel<<<NB / 8, 256, 0, stream>>>(logits, labels, accs + 0);
  gemm_softmax_kernel<<<NB / 32, 64, 0, stream>>>(feat, proto16, labels, accs + 1);
  gram_kernel<<<NT * NT, 32, 0, stream>>>(proto16, accs + 2);
  finalize_kernel<<<1, 1, 0, stream>>>(accs, out);
}
